// SelectiveSSM_11510512353488
// MI455X (gfx1250) — matrix operand order repaired and checked by reading, not compiled, not run on hardware
//
#include <hip/hip_runtime.h>
#include <hip/hip_bf16.h>

// ---------------------------------------------------------------------------
// Selective SSM (Mamba-style) pipeline for gfx1250 (MI455X).
// GEMMs: WMMA bf16 path with fp32 accumulation. B tiles are staged into LDS
// by the Tensor Data Mover (tensor_load_to_lds, TENSORcnt), double-buffered.
// Weights are transpose-converted to column-major bf16 once per launch.
// ---------------------------------------------------------------------------

typedef __attribute__((ext_vector_type(16))) __bf16 v16bf;
typedef __attribute__((ext_vector_type(8)))  __bf16 v8bf;
typedef __attribute__((ext_vector_type(4)))  __bf16 v4bf;
typedef __attribute__((ext_vector_type(8)))  float  v8f;
typedef __attribute__((ext_vector_type(4)))  unsigned int u32x4;
typedef __attribute__((ext_vector_type(8)))  int i32x8;
typedef __attribute__((ext_vector_type(4)))  int i32x4;

__device__ __forceinline__ __bf16 f2bf(float f) {
  union { float f; unsigned u; } v; v.f = f;
  unsigned u = v.u;
  u += 0x7FFFu + ((u >> 16) & 1u);          // round-to-nearest-even
  unsigned short h = (unsigned short)(u >> 16);
  union { unsigned short s; __bf16 b; } o; o.s = h;
  return o.b;
}

// A fragment (16x32 bf16): lane holds K [kb+16h, kb+16h+16) of row `row`
// -> two contiguous 16B global loads.
__device__ __forceinline__ v16bf load_a_frag(const __bf16* __restrict__ A,
                                             int row, int K, int kb, int half) {
  const __bf16* p = A + (size_t)row * K + kb + 16 * half;
  v8bf lo = *(const v8bf*)p;
  v8bf hi = *(const v8bf*)(p + 8);
  return __builtin_shufflevector(lo, hi, 0, 1, 2, 3, 4, 5, 6, 7,
                                 8, 9, 10, 11, 12, 13, 14, 15);
}

// B fragment from global (column-major Bt[N][K]): lane holds K [kb+16h,+16).
__device__ __forceinline__ v16bf load_b_frag(const __bf16* __restrict__ Bt,
                                             int col, int K, int kb, int half) {
  const __bf16* p = Bt + (size_t)col * K + kb + 16 * half;
  v8bf lo = *(const v8bf*)p;
  v8bf hi = *(const v8bf*)(p + 8);
  return __builtin_shufflevector(lo, hi, 0, 1, 2, 3, 4, 5, 6, 7,
                                 8, 9, 10, 11, 12, 13, 14, 15);
}

// B fragment from LDS-staged tile. TDM writes one 64B K-slab per column with
// 16B padding every 64B (pad_interval=16dw, pad_amount=4dw) -> 80B row stride,
// conflict-free ds_load_b128 across the 16 fragment lanes.
#define BTILE_ROW_STRIDE 80
#define BTILE_BYTES      (256 * BTILE_ROW_STRIDE)   // 20 KB per buffer

__device__ __forceinline__ v16bf load_b_frag_lds(const char* buf, int col_local,
                                                 int half) {
  const char* p = buf + col_local * BTILE_ROW_STRIDE + half * 32;
  v8bf lo = *(const v8bf*)p;
  v8bf hi = *(const v8bf*)(p + 16);
  return __builtin_shufflevector(lo, hi, 0, 1, 2, 3, 4, 5, 6, 7,
                                 8, 9, 10, 11, 12, 13, 14, 15);
}

#define WMMA_BF16(a, b, c) \
  __builtin_amdgcn_wmma_f32_16x16x32_bf16(false, (a), false, (b), (short)0, (c), false, false)

// ---------------------------------------------------------------------------
// TDM: DMA a 256(col) x 32(K) bf16 slab of Bt into LDS at lds_addr.
// D# per CDNA5 ISA ch.8: group0 = {count, lds_addr, global_addr, type=2},
// group1 = {data_size=2B, pad 4dw per 16dw, tensor_dim0=K, tensor_dim1=256,
//           tile_dim0=32, tile_dim1=256, tensor_dim0_stride=K}. 2D tile ->
// groups 2/3 zero.
// ---------------------------------------------------------------------------
__device__ __forceinline__ void tdm_load_b_tile(unsigned lds_addr,
                                                const __bf16* Bt,
                                                int nBase, int K, int kb) {
  const unsigned long long ga =
      (unsigned long long)(uintptr_t)(Bt + (size_t)nBase * K + kb);
  u32x4 g0;
  g0[0] = 1u;                                        // count=1 (valid)
  g0[1] = lds_addr;                                  // LDS byte address
  g0[2] = (unsigned)(ga & 0xFFFFFFFFu);              // global_addr[31:0]
  g0[3] = (unsigned)((ga >> 32) & 0x01FFFFFFu)       // global_addr[56:32]
          | (2u << 30);                              // type=2 (image)
  i32x8 g1;
  g1[0] = (1 << 16)                                  // data_size = 2 bytes
        | (1 << 20)                                  // pad_enable
        | (3 << 22)                                  // pad_interval: 16 dwords
        | (3 << 25);                                 // pad_amount: 4 dwords
  g1[1] = (K & 0xFFFF) << 16;                        // tensor_dim0[15:0]
  g1[2] = (K >> 16) | (256 << 16);                   // dim0[31:16] | dim1[15:0]
  g1[3] = (32 << 16);                                // dim1[31:16]=0 | tile_dim0
  g1[4] = 256;                                       // tile_dim1 (tile_dim2=0)
  g1[5] = K;                                         // tensor_dim0_stride lo
  g1[6] = 0;                                         // stride0 hi | stride1 lo
  g1[7] = 0;                                         // stride1 hi
  i32x4 z4 = {0, 0, 0, 0};
#if __clang_major__ >= 23
  i32x8 z8 = {0, 0, 0, 0, 0, 0, 0, 0};
  __builtin_amdgcn_tensor_load_to_lds(g0, g1, z4, z4, z8, 0);
#else
  __builtin_amdgcn_tensor_load_to_lds(g0, g1, z4, z4, 0);
#endif
}

// ---------------------------------------------------------------------------
// Big WMMA GEMM: C[M,N] = A[M,K] * Bt[N,K]^T. One 32x64 tile per wave
// (8 accumulators, 8 WMMAs per K-step). Block = 8 waves = 64x256 tile.
// B slab TDM-staged in LDS, double-buffered. Requires M%64==0, N%256==0,
// K%32==0. EPI: 0 plain, 1 softplus(x+bias[n]).
// ---------------------------------------------------------------------------
template <int EPI>
__global__ __launch_bounds__(256) void wmma_gemm_32x64(
    const __bf16* __restrict__ A, const __bf16* __restrict__ Bt,
    const float* __restrict__ bias, float* __restrict__ C,
    int M, int N, int K)
{
  __shared__ __align__(16) char smem[2 * BTILE_BYTES];

  const int lane = threadIdx.x & 31;
  const int wave = threadIdx.x >> 5;
  const int half = lane >> 4;
  const int l15  = lane & 15;
  const int mBlocks = M >> 6;
  const int bm = blockIdx.x % mBlocks;
  const int bn = blockIdx.x / mBlocks;
  const int mBase = bm * 64 + (wave >> 2) * 32;
  const int nBase = bn * 256 + (wave & 3) * 64;
  const int nBlockBase = bn * 256;

  const int rowA0 = mBase + l15;
  const int rowA1 = mBase + 16 + l15;
  const int colLocal0 = (wave & 3) * 64 + l15;

  const unsigned ldsBase = (unsigned)(uintptr_t)smem;   // low 32b = LDS offset

  v8f acc0 = {}, acc1 = {}, acc2 = {}, acc3 = {};
  v8f acc4 = {}, acc5 = {}, acc6 = {}, acc7 = {};

  if (wave == 0) tdm_load_b_tile(ldsBase, Bt, nBlockBase, K, 0);

  int cur = 0;
  for (int kb = 0; kb < K; kb += 32) {
    const bool more = (kb + 32) < K;
    if (wave == 0) {
      if (more) {
        tdm_load_b_tile(ldsBase + (cur ^ 1) * BTILE_BYTES, Bt, nBlockBase, K, kb + 32);
        __builtin_amdgcn_s_wait_tensorcnt((short)1);   // current buffer done
      } else {
        __builtin_amdgcn_s_wait_tensorcnt((short)0);   // last buffer done
      }
    }
    __syncthreads();                                   // publish staged B slab

    __builtin_prefetch(A + (size_t)rowA0 * K + kb + 32, 0, 1);

    const char* bufp = smem + cur * BTILE_BYTES;
    const v16bf a0 = load_a_frag(A, rowA0, K, kb, half);
    const v16bf a1 = load_a_frag(A, rowA1, K, kb, half);
    const v16bf b0 = load_b_frag_lds(bufp, colLocal0 +  0, half);
    const v16bf b1 = load_b_frag_lds(bufp, colLocal0 + 16, half);
    const v16bf b2 = load_b_frag_lds(bufp, colLocal0 + 32, half);
    const v16bf b3 = load_b_frag_lds(bufp, colLocal0 + 48, half);

    acc0 = WMMA_BF16(a0, b0, acc0);
    acc1 = WMMA_BF16(a0, b1, acc1);
    acc2 = WMMA_BF16(a0, b2, acc2);
    acc3 = WMMA_BF16(a0, b3, acc3);
    acc4 = WMMA_BF16(a1, b0, acc4);
    acc5 = WMMA_BF16(a1, b1, acc5);
    acc6 = WMMA_BF16(a1, b2, acc6);
    acc7 = WMMA_BF16(a1, b3, acc7);

    __syncthreads();                                   // buffer reuse guard
    cur ^= 1;
  }

  v8f accs[8] = {acc0, acc1, acc2, acc3, acc4, acc5, acc6, acc7};
  #pragma unroll
  for (int mi = 0; mi < 2; ++mi) {
    #pragma unroll
    for (int nt = 0; nt < 4; ++nt) {
      const int n = nBase + nt * 16 + l15;
      #pragma unroll
      for (int r = 0; r < 8; ++r) {
        const int m = mBase + mi * 16 + r + 8 * half;
        float v = accs[mi * 4 + nt][r];
        if (EPI == 1) {
          v += bias[n];
          v = (v > 20.0f) ? v : log1pf(__expf(v));     // softplus
        }
        C[(size_t)m * N + n] = v;
      }
    }
  }
}

// ---------------------------------------------------------------------------
// Small-N WMMA GEMM (N==16): one 16x16 tile per wave. Used for Bs = xs @ W_x.
// ---------------------------------------------------------------------------
__global__ __launch_bounds__(256) void wmma_gemm_n16(
    const __bf16* __restrict__ A, const __bf16* __restrict__ Bt,
    float* __restrict__ C, int M, int K)
{
  const int lane = threadIdx.x & 31;
  const int wave = threadIdx.x >> 5;
  const int tile = blockIdx.x * 8 + wave;        // uniform per wave
  if (tile >= (M >> 4)) return;                  // whole-wave exit, EXEC all-1
  const int half = lane >> 4;
  const int l15  = lane & 15;
  const int rowA = tile * 16 + l15;

  v8f acc = {};
  for (int kb = 0; kb < K; kb += 32) {
    const v16bf a = load_a_frag(A, rowA, K, kb, half);
    const v16bf b = load_b_frag(Bt, l15, K, kb, half);
    acc = WMMA_BF16(a, b, acc);
  }
  #pragma unroll
  for (int r = 0; r < 8; ++r) {
    const int m = tile * 16 + r + 8 * half;
    C[(size_t)m * 16 + l15] = acc[r];
  }
}

// ---------------------------------------------------------------------------
// Elementwise fp32 -> bf16 (4 elements / thread, n % 4 == 0).
// ---------------------------------------------------------------------------
__global__ __launch_bounds__(256) void convert_bf16_kernel(
    const float* __restrict__ src, __bf16* __restrict__ dst, size_t n4)
{
  const size_t i = (size_t)blockIdx.x * blockDim.x + threadIdx.x;
  if (i >= n4) return;
  const float4 f = ((const float4*)src)[i];
  v4bf o;
  o[0] = f2bf(f.x); o[1] = f2bf(f.y); o[2] = f2bf(f.z); o[3] = f2bf(f.w);
  ((v4bf*)dst)[i] = o;
}

// ---------------------------------------------------------------------------
// Transpose-convert: src[K,N] f32 row-major -> dst[N,K] bf16 (column-major B).
// ---------------------------------------------------------------------------
__global__ __launch_bounds__(256) void transpose_bf16_kernel(
    const float* __restrict__ src, __bf16* __restrict__ dst, int K, int N)
{
  const size_t i = (size_t)blockIdx.x * blockDim.x + threadIdx.x;
  if (i >= (size_t)K * N) return;
  const int k = (int)(i % K);
  const int n = (int)(i / K);
  dst[i] = f2bf(src[(size_t)k * N + n]);
}

// ---------------------------------------------------------------------------
// Depthwise causal conv (K=4) + bias + SiLU; writes f32 (for scan) and bf16
// (A-matrix of the dt / Bs GEMMs).
// ---------------------------------------------------------------------------
__global__ __launch_bounds__(256) void conv_silu_kernel(
    const float* __restrict__ xz, const float* __restrict__ conv_w,
    const float* __restrict__ conv_b, float* __restrict__ xs_f32,
    __bf16* __restrict__ xs_bf, int L, int Din, int Dxz)
{
  const size_t idx = (size_t)blockIdx.x * blockDim.x + threadIdx.x;
  const int d = (int)(idx % Din);
  const size_t bt = idx / Din;                    // b*L + t
  const int t = (int)(bt % L);
  const size_t bL = bt - t;
  float acc = conv_b[d];
  const float* w = conv_w + (size_t)d * 4;
  #pragma unroll
  for (int k = 0; k < 4; ++k) {
    const int tt = t - 3 + k;
    if (tt >= 0) acc += w[k] * xz[(bL + tt) * Dxz + d];
  }
  const float s = acc / (1.0f + __expf(-acc));    // silu
  xs_f32[idx] = s;
  xs_bf[idx]  = f2bf(s);
}

// ---------------------------------------------------------------------------
// Selective scan. 16 lanes per (b,d) channel, one state per lane.
//   h_s(t) = h_s(t-1)*exp(-A_s*dt) + Bs_s(t);  y = (sum_s h_s + D*x)*silu(z)
// Writes y directly as bf16 (A-matrix of the output GEMM).
// ---------------------------------------------------------------------------
__global__ __launch_bounds__(256) void scan_kernel(
    const float* __restrict__ dtbuf, const float* __restrict__ Bsbuf,
    const float* __restrict__ xs_f32, const float* __restrict__ xz,
    const float* __restrict__ A_log, const float* __restrict__ Dvec,
    __bf16* __restrict__ y_bf, int Bn, int L, int Din)
{
  const int g  = blockIdx.x * blockDim.x + threadIdx.x;
  const int s  = g & 15;
  const int ch = g >> 4;
  if (ch >= Bn * Din) return;
  const int d = ch % Din;
  const int b = ch / Din;

  const float Aexp = __expf(A_log[d * 16 + s]);
  const float Dd   = Dvec[d];
  float h = 0.0f;

  for (int t = 0; t < L; ++t) {
    const size_t row = (size_t)(b * L + t);
    const float dtv = dtbuf[row * Din + d];
    const float bsv = Bsbuf[row * 16 + s];
    h = h * __expf(-Aexp * dtv) + bsv;
    float p = h;                                  // reduce over 16 states
    p += __shfl_xor(p, 1, 16);
    p += __shfl_xor(p, 2, 16);
    p += __shfl_xor(p, 4, 16);
    p += __shfl_xor(p, 8, 16);
    const float xv = xs_f32[row * Din + d];
    const float zv = xz[row * (size_t)(2 * Din) + Din + d];
    float y = p + Dd * xv;
    y *= zv / (1.0f + __expf(-zv));               // * silu(z)
    if (s == 0) y_bf[row * Din + d] = f2bf(y);
  }
}

// ---------------------------------------------------------------------------
extern "C" void kernel_launch(void* const* d_in, const int* in_sizes, int n_in,
                              void* d_out, int out_size, void* d_ws, size_t ws_size,
                              hipStream_t stream)
{
  const float* x      = (const float*)d_in[0];
  const float* W_in   = (const float*)d_in[1];
  const float* conv_w = (const float*)d_in[2];
  const float* conv_b = (const float*)d_in[3];
  const float* A_log  = (const float*)d_in[4];
  const float* Dvec   = (const float*)d_in[5];
  const float* W_x    = (const float*)d_in[6];
  const float* W_dt   = (const float*)d_in[7];
  const float* b_dt   = (const float*)d_in[8];
  const float* W_out  = (const float*)d_in[9];
  float* out = (float*)d_out;

  const int Bn = 4, L = 4096, Dm = 1024, Din = 2048, S = 16;
  const int Mrows = Bn * L;                       // 16384
  const int Nxz = 2 * Din;                        // 4096

  // ---- workspace layout (256B-aligned regions) ----
  char* ws = (char*)d_ws;
  size_t off = 0;
  auto carve = [&](size_t bytes) {
    size_t r = off;
    off = (off + bytes + 255) & ~(size_t)255;
    return r;
  };
  float*  xz      = (float*) (ws + carve((size_t)Mrows * Nxz * 4));
  float*  xs_f32  = (float*) (ws + carve((size_t)Mrows * Din * 4));
  float*  dtb     = (float*) (ws + carve((size_t)Mrows * Din * 4));
  float*  Bsb     = (float*) (ws + carve((size_t)Mrows * S * 4));
  __bf16* x_bf    = (__bf16*)(ws + carve((size_t)Mrows * Dm * 2));
  __bf16* xs_bf   = (__bf16*)(ws + carve((size_t)Mrows * Din * 2));
  __bf16* y_bf    = (__bf16*)(ws + carve((size_t)Mrows * Din * 2));
  __bf16* WinT    = (__bf16*)(ws + carve((size_t)Nxz * Dm * 2));
  __bf16* WdtT    = (__bf16*)(ws + carve((size_t)Din * Din * 2));
  __bf16* WxT     = (__bf16*)(ws + carve((size_t)S * Din * 2));
  __bf16* WoutT   = (__bf16*)(ws + carve((size_t)Dm * Din * 2));

  // ---- one-time conversions / weight transposes ----
  {
    size_t n4 = (size_t)Mrows * Dm / 4;
    convert_bf16_kernel<<<(unsigned)((n4 + 255) / 256), 256, 0, stream>>>(x, x_bf, n4);
  }
  {
    size_t n = (size_t)Dm * Nxz;
    transpose_bf16_kernel<<<(unsigned)((n + 255) / 256), 256, 0, stream>>>(W_in, WinT, Dm, Nxz);
  }
  {
    size_t n = (size_t)Din * Din;
    transpose_bf16_kernel<<<(unsigned)((n + 255) / 256), 256, 0, stream>>>(W_dt, WdtT, Din, Din);
  }
  {
    size_t n = (size_t)Din * S;
    transpose_bf16_kernel<<<(unsigned)((n + 255) / 256), 256, 0, stream>>>(W_x, WxT, Din, S);
  }
  {
    size_t n = (size_t)Din * Dm;
    transpose_bf16_kernel<<<(unsigned)((n + 255) / 256), 256, 0, stream>>>(W_out, WoutT, Din, Dm);
  }

  // 1) xz = x @ W_in : M=16384, N=4096, K=1024
  wmma_gemm_32x64<0><<<(Mrows / 64) * (Nxz / 256), 256, 0, stream>>>(
      x_bf, WinT, nullptr, xz, Mrows, Nxz, Dm);

  // 2) conv + silu -> xs (f32 + bf16)
  {
    size_t total = (size_t)Mrows * Din;
    conv_silu_kernel<<<(unsigned)((total + 255) / 256), 256, 0, stream>>>(
        xz, conv_w, conv_b, xs_f32, xs_bf, L, Din, Nxz);
  }

  // 3) dt = softplus(xs @ W_dt + b_dt) : M=16384, N=2048, K=2048
  wmma_gemm_32x64<1><<<(Mrows / 64) * (Din / 256), 256, 0, stream>>>(
      xs_bf, WdtT, b_dt, dtb, Mrows, Din, Din);

  // 4) Bs = xs @ W_x : M=16384, N=16, K=2048
  wmma_gemm_n16<<<(Mrows / 16 + 7) / 8, 256, 0, stream>>>(
      xs_bf, WxT, Bsb, Mrows, Din);

  // 5) selective scan -> y (bf16)
  scan_kernel<<<(Bn * Din * 16) / 256, 256, 0, stream>>>(
      dtb, Bsb, xs_f32, xz, A_log, Dvec, y_bf, Bn, L, Din);

  // 6) out = y @ W_out : M=16384, N=1024, K=2048
  wmma_gemm_32x64<0><<<(Mrows / 64) * (Dm / 256), 256, 0, stream>>>(
      y_bf, WoutT, nullptr, out, Mrows, Dm, Din);
}
